// MAGIKNet_72653666779783
// MI455X (gfx1250) — matrix operand order repaired and checked by reading, not compiled, not run on hardware
//
#include <hip/hip_runtime.h>
#include <hip/hip_bf16.h>

#define N_NODES 50000
#define N_EDGES 800000
#define HID     128
#define BN_EPS  1e-5f
#define STATS_BLOCKS 1024

typedef _Float16 hvec8 __attribute__((ext_vector_type(8)));
typedef _Float16 v16h  __attribute__((ext_vector_type(16)));
typedef float    v8f   __attribute__((ext_vector_type(8)));

// ---------------------------------------------------------------------------
// WMMA helpers (CDNA5: V_WMMA_F32_16X16X32_F16, wave32)
// ---------------------------------------------------------------------------
__device__ __forceinline__ v8f wmma_f16(v16h a, v16h b, v8f c) {
  // 8 args: (neg_a, A, neg_b, B, c_mod, C, reuse_a, reuse_b)
  return __builtin_amdgcn_wmma_f32_16x16x32_f16(false, a, false, b, (short)0, c,
                                                false, false);
}

// A-matrix 16x32 f16 fragment. Lane group g = lane>>4 holds:
//   a[0..7]  = A[row][g*16 + 0..7]       (VGPR0-3)
//   a[8..15] = A[row][g*16 + 8 + 0..7]   (VGPR4-7)
// p points at A[row][0] of a 32-wide K slice. Two contiguous 16B loads.
__device__ __forceinline__ v16h load_a32(const _Float16* __restrict__ p, int g) {
  hvec8 lo = *(const hvec8*)(p + g * 16);
  hvec8 hi = *(const hvec8*)(p + g * 16 + 8);
  v16h a;
#pragma unroll
  for (int i = 0; i < 8; ++i) { a[i] = lo[i]; a[8 + i] = hi[i]; }
  return a;
}

// B-matrix 32x16 f16 fragment per K-step: lane column = lane&15,
// b[k'] = W[step*32 + g*16 + k'][col]. Loaded once per wave, reused all tiles.
template <int KSTEPS, int NC>
__device__ __forceinline__ void load_b_frags(const _Float16* __restrict__ W,
                                             int col, int g, v16h (&bf)[KSTEPS]) {
#pragma unroll
  for (int s = 0; s < KSTEPS; ++s) {
#pragma unroll
    for (int k = 0; k < 16; ++k)
      bf[s][k] = W[(size_t)(s * 32 + g * 16 + k) * NC + col];
  }
}

// ---------------------------------------------------------------------------
// GEMM 1: conv message layer.  out[e][:] = concat(xh[dst[e]], xh[src[e]]) @ W + b
// K = 256 (8 steps), NC = 128, block = 256 threads (8 waves, one 16-col strip each)
// ---------------------------------------------------------------------------
__global__ void gemm_conv1(const _Float16* __restrict__ xh,
                           const int* __restrict__ eidx,   // [2*E]: src, then dst
                           const _Float16* __restrict__ Wh, // 256x128 f16
                           const float* __restrict__ bias,
                           float* __restrict__ out) {
  const int lane = threadIdx.x & 31;
  const int wave = threadIdx.x >> 5;
  const int g = lane >> 4;
  const int ln = lane & 15;
  const int col = wave * 16 + ln;

  v16h bf[8];
  load_b_frags<8, HID>(Wh, col, g, bf);
  const float bv = bias[col];

  const int ntiles = N_EDGES / 16;
  for (int tile = blockIdx.x; tile < ntiles; tile += gridDim.x) {
    const int row = tile * 16 + ln;
    const int si = eidx[row];
    const int di = eidx[N_EDGES + row];
    const _Float16* pd = xh + (size_t)di * HID;
    const _Float16* ps = xh + (size_t)si * HID;
    v8f c = {};
#pragma unroll
    for (int s = 0; s < 4; ++s) c = wmma_f16(load_a32(pd + s * 32, g), bf[s], c);
#pragma unroll
    for (int s = 0; s < 4; ++s) c = wmma_f16(load_a32(ps + s * 32, g), bf[4 + s], c);
    float* o = out + (size_t)tile * 16 * HID + col;
#pragma unroll
    for (int j = 0; j < 8; ++j) o[(size_t)(j + 8 * g) * HID] = c[j] + bv;
  }
}

// ---------------------------------------------------------------------------
// GEMM 2: classifier layer 1. A = concat(xh[src], xh[dst], efh[e]) (K padded to 288)
// ---------------------------------------------------------------------------
__global__ void gemm_cls1(const _Float16* __restrict__ xh,
                          const _Float16* __restrict__ efh, // E x 32 (4 real + pad)
                          const int* __restrict__ eidx,
                          const _Float16* __restrict__ Wh, // 288x128 (rows 260+ zero)
                          const float* __restrict__ bias,
                          float* __restrict__ out) {
  const int lane = threadIdx.x & 31;
  const int wave = threadIdx.x >> 5;
  const int g = lane >> 4;
  const int ln = lane & 15;
  const int col = wave * 16 + ln;

  v16h bf[9];
  load_b_frags<9, HID>(Wh, col, g, bf);
  const float bv = bias[col];

  const int ntiles = N_EDGES / 16;
  for (int tile = blockIdx.x; tile < ntiles; tile += gridDim.x) {
    const int row = tile * 16 + ln;
    const int si = eidx[row];
    const int di = eidx[N_EDGES + row];
    const _Float16* ps = xh + (size_t)si * HID;
    const _Float16* pd = xh + (size_t)di * HID;
    const _Float16* pe = efh + (size_t)row * 32;
    v8f c = {};
#pragma unroll
    for (int s = 0; s < 4; ++s) c = wmma_f16(load_a32(ps + s * 32, g), bf[s], c);
#pragma unroll
    for (int s = 0; s < 4; ++s) c = wmma_f16(load_a32(pd + s * 32, g), bf[4 + s], c);
    c = wmma_f16(load_a32(pe, g), bf[8], c);
    float* o = out + (size_t)tile * 16 * HID + col;
#pragma unroll
    for (int j = 0; j < 8; ++j) o[(size_t)(j + 8 * g) * HID] = c[j] + bv;
  }
}

// ---------------------------------------------------------------------------
// GEMM 3: dense rows. A is rows x (KSTEPS*32) f16, W is K x NC.
// blockDim must be NC*2 (NC/16 waves).
// ---------------------------------------------------------------------------
template <int KSTEPS, int NC>
__global__ void gemm_dense(const _Float16* __restrict__ A,
                           const _Float16* __restrict__ Wh,
                           const float* __restrict__ bias,
                           float* __restrict__ out, int rows) {
  const int lane = threadIdx.x & 31;
  const int wave = threadIdx.x >> 5;
  const int g = lane >> 4;
  const int ln = lane & 15;
  const int col = wave * 16 + ln;
  const int K = KSTEPS * 32;

  v16h bf[KSTEPS];
  load_b_frags<KSTEPS, NC>(Wh, col, g, bf);
  const float bv = bias[col];

  const int ntiles = rows / 16;
  for (int tile = blockIdx.x; tile < ntiles; tile += gridDim.x) {
    const _Float16* pa = A + (size_t)(tile * 16 + ln) * K;
    v8f c = {};
#pragma unroll
    for (int s = 0; s < KSTEPS; ++s) c = wmma_f16(load_a32(pa + s * 32, g), bf[s], c);
    float* o = out + (size_t)tile * 16 * NC + col;
#pragma unroll
    for (int j = 0; j < 8; ++j) o[(size_t)(j + 8 * g) * NC] = c[j] + bv;
  }
}

// ---------------------------------------------------------------------------
// Deterministic batch-norm statistics: two-stage column reduction.
// blockDim fixed at 256. Stage 1: per-block partial sums (LDS sub-reduce).
// Stage 2: fixed-order reduce over STATS_BLOCKS partials.
// ---------------------------------------------------------------------------
__global__ void col_stats_part(const float* __restrict__ x, int rows, int C,
                               float* __restrict__ part) {
  const int c = threadIdx.x % C;
  const int rper = 256 / C;
  const int sub = threadIdx.x / C;
  float s = 0.f, q = 0.f;
  for (int r = blockIdx.x * rper + sub; r < rows; r += gridDim.x * rper) {
    float v = x[(size_t)r * C + c];
    s += v;
    q += v * v;
  }
  __shared__ float shs[256];
  __shared__ float shq[256];
  shs[threadIdx.x] = s;
  shq[threadIdx.x] = q;
  __syncthreads();
  if (sub == 0) {
    for (int u = 1; u < rper; ++u) { s += shs[u * C + c]; q += shq[u * C + c]; }
    part[(size_t)blockIdx.x * 2 * C + c] = s;
    part[(size_t)blockIdx.x * 2 * C + C + c] = q;
  }
}

__global__ void col_stats_reduce(const float* __restrict__ part, int C,
                                 float* __restrict__ sums) {
  const int t = threadIdx.x;
  if (t < 2 * C) {
    float s = 0.f;
    for (int b = 0; b < STATS_BLOCKS; ++b) s += part[(size_t)b * 2 * C + t];
    sums[t] = s;
  }
}

__global__ void bn_finalize(const float* __restrict__ sums,
                            const float* __restrict__ gamma,
                            const float* __restrict__ beta, float rows, int C,
                            float* __restrict__ scale, float* __restrict__ shift) {
  const int c = threadIdx.x;
  if (c < C) {
    const float mu = sums[c] / rows;
    const float var = sums[C + c] / rows - mu * mu;
    const float sc = gamma[c] * rsqrtf(var + BN_EPS);
    scale[c] = sc;
    shift[c] = beta[c] - mu * sc;
  }
}

// ---------------------------------------------------------------------------
// Elementwise BN-apply / ReLU / conversion kernels
// ---------------------------------------------------------------------------
__global__ void bn_relu_f16(const float* __restrict__ x, const float* __restrict__ scale,
                            const float* __restrict__ shift, _Float16* __restrict__ y,
                            size_t n, int cmask) {
  size_t i = (size_t)blockIdx.x * blockDim.x + threadIdx.x;
  if (i < n) {
    const int c = (int)(i & (size_t)cmask);
    float v = fmaf(x[i], scale[c], shift[c]);
    y[i] = (_Float16)(v > 0.f ? v : 0.f);
  }
}

__global__ void bn_relu_f32f16(const float* __restrict__ x, const float* __restrict__ scale,
                               const float* __restrict__ shift, float* __restrict__ yf,
                               _Float16* __restrict__ yh, size_t n, int cmask) {
  size_t i = (size_t)blockIdx.x * blockDim.x + threadIdx.x;
  if (i < n) {
    const int c = (int)(i & (size_t)cmask);
    float v = fmaf(x[i], scale[c], shift[c]);
    v = v > 0.f ? v : 0.f;
    yf[i] = v;
    yh[i] = (_Float16)v;
  }
}

__global__ void bn_relu_f32_inplace(float* __restrict__ x, const float* __restrict__ scale,
                                    const float* __restrict__ shift, size_t n, int cmask) {
  size_t i = (size_t)blockIdx.x * blockDim.x + threadIdx.x;
  if (i < n) {
    const int c = (int)(i & (size_t)cmask);
    float v = fmaf(x[i], scale[c], shift[c]);
    x[i] = v > 0.f ? v : 0.f;
  }
}

// segment_max over dst with ReLU'd (>=0) values: uint atomicMax on float bits,
// agg pre-zeroed -> isolated nodes get 0 exactly like the jnp.where(isfinite) path.
__global__ void bn_relu_segmax(const float* __restrict__ m, const float* __restrict__ scale,
                               const float* __restrict__ shift, const int* __restrict__ eidx,
                               float* __restrict__ agg, size_t n) {
  size_t i = (size_t)blockIdx.x * blockDim.x + threadIdx.x;
  if (i < n) {
    const int c = (int)(i & (size_t)(HID - 1));
    const size_t e = i >> 7;
    const int d = eidx[N_EDGES + e];
    float v = fmaf(m[i], scale[c], shift[c]);
    v = v > 0.f ? v : 0.f;
    atomicMax((unsigned int*)&agg[(size_t)d * HID + c], __float_as_uint(v));
  }
}

__global__ void residual_add(const float* __restrict__ agg, float* __restrict__ xf,
                             _Float16* __restrict__ xh, size_t n) {
  size_t i = (size_t)blockIdx.x * blockDim.x + threadIdx.x;
  if (i < n) {
    const float v = xf[i] + agg[i];
    xf[i] = v;
    xh[i] = (_Float16)v;
  }
}

__global__ void zero_f32(float* __restrict__ p, size_t n) {
  size_t i = (size_t)blockIdx.x * blockDim.x + threadIdx.x;
  if (i < n) p[i] = 0.f;
}

// ---------------------------------------------------------------------------
// Small encoders / converters / final head
// ---------------------------------------------------------------------------
__global__ void node_enc_l1(const float* __restrict__ nf, const float* __restrict__ W,
                            const float* __restrict__ b, float* __restrict__ out) {
  const int i = blockIdx.x * blockDim.x + threadIdx.x;
  if (i < N_NODES * HID) {
    const int c = i & (HID - 1);
    const int n = i >> 7;
    float acc = b[c];
#pragma unroll
    for (int k = 0; k < 6; ++k) acc = fmaf(nf[n * 6 + k], W[k * HID + c], acc);
    out[i] = acc;
  }
}

__global__ void cvt_f32_to_f16(const float* __restrict__ src, _Float16* __restrict__ dst,
                               int n) {
  const int i = blockIdx.x * blockDim.x + threadIdx.x;
  if (i < n) dst[i] = (_Float16)src[i];
}

__global__ void cvt_clsW1(const float* __restrict__ src /*260x128*/,
                          _Float16* __restrict__ dst /*288x128*/) {
  const int i = blockIdx.x * blockDim.x + threadIdx.x;
  if (i < 288 * HID) {
    const int r = i >> 7;
    dst[i] = (r < 260) ? (_Float16)src[i] : (_Float16)0.f;
  }
}

__global__ void pad_edge_feats(const float* __restrict__ ef, _Float16* __restrict__ efh) {
  const int i = blockIdx.x * blockDim.x + threadIdx.x;
  if (i < N_EDGES * 32) {
    const int c = i & 31;
    const int e = i >> 5;
    efh[i] = (c < 4) ? (_Float16)ef[e * 4 + c] : (_Float16)0.f;
  }
}

__global__ void cls_l3_sigmoid(const float* __restrict__ h /*E x 64*/,
                               const float* __restrict__ W3, const float* __restrict__ b3,
                               float* __restrict__ out) {
  const int e = blockIdx.x * blockDim.x + threadIdx.x;
  if (e < N_EDGES) {
    float acc = b3[0];
#pragma unroll
    for (int k = 0; k < 64; ++k) acc = fmaf(h[(size_t)e * 64 + k], W3[k], acc);
    out[e] = 1.f / (1.f + __expf(-acc));
  }
}

// ---------------------------------------------------------------------------
// Host orchestration
// ---------------------------------------------------------------------------
static inline int cdiv_i(size_t a, int b) { return (int)((a + (size_t)b - 1) / (size_t)b); }

extern "C" void kernel_launch(void* const* d_in, const int* in_sizes, int n_in,
                              void* d_out, int out_size, void* d_ws, size_t ws_size,
                              hipStream_t stream) {
  (void)in_sizes; (void)n_in; (void)out_size; (void)ws_size;

  // jax pytree (sorted-key) flatten order of setup_inputs()
  const float* ef   = (const float*)d_in[0];   // edge_features (E,4)
  const int*   eidx = (const int*)d_in[1];     // edge_index (2,E) int32
  const float* nf   = (const float*)d_in[2];   // node_features (N,6)
  // cls: bn1.b, bn1.g, bn2.b, bn2.g, l1.W, l1.b, l2.W, l2.b, l3.W, l3.b
  const float* cls_bn1_b = (const float*)d_in[3];
  const float* cls_bn1_g = (const float*)d_in[4];
  const float* cls_bn2_b = (const float*)d_in[5];
  const float* cls_bn2_g = (const float*)d_in[6];
  const float* cls_l1_W  = (const float*)d_in[7];
  const float* cls_l1_b  = (const float*)d_in[8];
  const float* cls_l2_W  = (const float*)d_in[9];
  const float* cls_l2_b  = (const float*)d_in[10];
  const float* cls_l3_W  = (const float*)d_in[11];
  const float* cls_l3_b  = (const float*)d_in[12];
  // convs[l]: bn1.b, bn1.g, bn2.b, bn2.g, l1.W, l1.b, l2.W, l2.b  (base 13, stride 8)
  auto conv_ptr = [&](int l, int k) { return (const float*)d_in[13 + l * 8 + k]; };
  // d_in[37..40] = edge_enc (computed-but-unused in reference): skipped
  const float* ne_bn1_b = (const float*)d_in[41];
  const float* ne_bn1_g = (const float*)d_in[42];
  const float* ne_bn2_b = (const float*)d_in[43];
  const float* ne_bn2_g = (const float*)d_in[44];
  const float* ne_l1_W  = (const float*)d_in[45];
  const float* ne_l1_b  = (const float*)d_in[46];
  const float* ne_l2_W  = (const float*)d_in[47];
  const float* ne_l2_b  = (const float*)d_in[48];

  float* out = (float*)d_out;

  // ---- workspace carve-up (256B aligned) ----
  size_t off = 0;
  auto take = [&](size_t bytes) {
    size_t o = off;
    off += (bytes + 255) & ~(size_t)255;
    return o;
  };
  char* ws = (char*)d_ws;
  float*    sums  = (float*)(ws + take(2 * HID * sizeof(float)));
  float*    scale = (float*)(ws + take(HID * sizeof(float)));
  float*    shift = (float*)(ws + take(HID * sizeof(float)));
  float*    part  = (float*)(ws + take((size_t)STATS_BLOCKS * 2 * HID * sizeof(float)));
  _Float16* xh    = (_Float16*)(ws + take((size_t)N_NODES * HID * 2));
  float*    xf    = (float*)(ws + take((size_t)N_NODES * HID * 4));
  float*    agg   = (float*)(ws + take((size_t)N_NODES * HID * 4));
  float*    bufA  = (float*)(ws + take((size_t)N_EDGES * HID * 4));
  _Float16* bufH  = (_Float16*)(ws + take((size_t)N_EDGES * HID * 2));
  _Float16* efh   = (_Float16*)(ws + take((size_t)N_EDGES * 32 * 2));
  _Float16* convW1h[3];
  _Float16* convW2h[3];
  for (int l = 0; l < 3; ++l) {
    convW1h[l] = (_Float16*)(ws + take((size_t)256 * HID * 2));
    convW2h[l] = (_Float16*)(ws + take((size_t)HID * HID * 2));
  }
  _Float16* neW2h  = (_Float16*)(ws + take((size_t)HID * HID * 2));
  _Float16* clsW1h = (_Float16*)(ws + take((size_t)288 * HID * 2));
  _Float16* clsW2h = (_Float16*)(ws + take((size_t)HID * 64 * 2));

  const int EW = 256;  // elementwise block
  auto run_stats = [&](const float* x, int rows, int C) {
    col_stats_part<<<STATS_BLOCKS, 256, 0, stream>>>(x, rows, C, part);
    col_stats_reduce<<<1, 256, 0, stream>>>(part, C, sums);
  };

  // ---- weight prep (f32 -> f16, padding) ----
  for (int l = 0; l < 3; ++l) {
    cvt_f32_to_f16<<<cdiv_i(256 * HID, EW), EW, 0, stream>>>(conv_ptr(l, 4), convW1h[l], 256 * HID);
    cvt_f32_to_f16<<<cdiv_i(HID * HID, EW), EW, 0, stream>>>(conv_ptr(l, 6), convW2h[l], HID * HID);
  }
  cvt_f32_to_f16<<<cdiv_i(HID * HID, EW), EW, 0, stream>>>(ne_l2_W, neW2h, HID * HID);
  cvt_clsW1<<<cdiv_i(288 * HID, EW), EW, 0, stream>>>(cls_l1_W, clsW1h);
  cvt_f32_to_f16<<<cdiv_i(HID * 64, EW), EW, 0, stream>>>(cls_l2_W, clsW2h, HID * 64);
  pad_edge_feats<<<cdiv_i((size_t)N_EDGES * 32, EW), EW, 0, stream>>>(ef, efh);

  const size_t nN = (size_t)N_NODES * HID;
  const size_t nE = (size_t)N_EDGES * HID;

  // ---- node encoder ----
  node_enc_l1<<<cdiv_i(nN, EW), EW, 0, stream>>>(nf, ne_l1_W, ne_l1_b, bufA);
  run_stats(bufA, N_NODES, HID);
  bn_finalize<<<1, HID, 0, stream>>>(sums, ne_bn1_g, ne_bn1_b, (float)N_NODES, HID, scale, shift);
  bn_relu_f16<<<cdiv_i(nN, EW), EW, 0, stream>>>(bufA, scale, shift, xh, nN, HID - 1);

  gemm_dense<4, HID><<<1024, 256, 0, stream>>>(xh, neW2h, ne_l2_b, bufA, N_NODES);
  run_stats(bufA, N_NODES, HID);
  bn_finalize<<<1, HID, 0, stream>>>(sums, ne_bn2_g, ne_bn2_b, (float)N_NODES, HID, scale, shift);
  bn_relu_f32f16<<<cdiv_i(nN, EW), EW, 0, stream>>>(bufA, scale, shift, xf, xh, nN, HID - 1);

  // ---- conv layers ----
  for (int l = 0; l < 3; ++l) {
    gemm_conv1<<<2048, 256, 0, stream>>>(xh, eidx, convW1h[l], conv_ptr(l, 5), bufA);
    run_stats(bufA, N_EDGES, HID);
    bn_finalize<<<1, HID, 0, stream>>>(sums, conv_ptr(l, 1), conv_ptr(l, 0),
                                       (float)N_EDGES, HID, scale, shift);
    bn_relu_f16<<<cdiv_i(nE, EW), EW, 0, stream>>>(bufA, scale, shift, bufH, nE, HID - 1);

    gemm_dense<4, HID><<<2048, 256, 0, stream>>>(bufH, convW2h[l], conv_ptr(l, 7), bufA, N_EDGES);
    run_stats(bufA, N_EDGES, HID);
    bn_finalize<<<1, HID, 0, stream>>>(sums, conv_ptr(l, 3), conv_ptr(l, 2),
                                       (float)N_EDGES, HID, scale, shift);

    zero_f32<<<cdiv_i(nN, EW), EW, 0, stream>>>(agg, nN);
    bn_relu_segmax<<<cdiv_i(nE, EW), EW, 0, stream>>>(bufA, scale, shift, eidx, agg, nE);
    residual_add<<<cdiv_i(nN, EW), EW, 0, stream>>>(agg, xf, xh, nN);
  }

  // ---- classifier ----
  gemm_cls1<<<2048, 256, 0, stream>>>(xh, efh, eidx, clsW1h, cls_l1_b, bufA);
  run_stats(bufA, N_EDGES, HID);
  bn_finalize<<<1, HID, 0, stream>>>(sums, cls_bn1_g, cls_bn1_b, (float)N_EDGES, HID, scale, shift);
  bn_relu_f16<<<cdiv_i(nE, EW), EW, 0, stream>>>(bufA, scale, shift, bufH, nE, HID - 1);

  gemm_dense<4, 64><<<2048, 128, 0, stream>>>(bufH, clsW2h, cls_l2_b, bufA, N_EDGES);
  run_stats(bufA, N_EDGES, 64);
  bn_finalize<<<1, 64, 0, stream>>>(sums, cls_bn2_g, cls_bn2_b, (float)N_EDGES, 64, scale, shift);
  const size_t nE64 = (size_t)N_EDGES * 64;
  bn_relu_f32_inplace<<<cdiv_i(nE64, EW), EW, 0, stream>>>(bufA, scale, shift, nE64, 63);

  cls_l3_sigmoid<<<cdiv_i(N_EDGES, EW), EW, 0, stream>>>(bufA, cls_l3_W, cls_l3_b, out);
}
